// ResidualFlow_DiffEmbTransformer_74972949119279
// MI455X (gfx1250) — compile-verified, matrix operand order checked by reading, not run on hardware
//
#include <hip/hip_runtime.h>
#include <hip/hip_bf16.h>
#include <math.h>

// ---------------------------------------------------------------------------
// CDNA5 / gfx1250 implementation. wave32. Matrix ops: v_wmma_f32_16x16x32_bf16.
// ---------------------------------------------------------------------------

typedef __bf16 bf16_t;
typedef __attribute__((ext_vector_type(16))) __bf16 v16bf;
typedef __attribute__((ext_vector_type(8)))  __bf16 v8bf;
typedef __attribute__((ext_vector_type(8)))  float  v8f;

#define B_   2
#define N_   256
#define F_   512
#define H1P  304   // H1=300 padded to multiple of 16
#define H1K  320   // K padding for the tail GEMM (multiple of 32)
#define H2P  112   // H2=100 padded to multiple of 16

__device__ __forceinline__ v8f wmma_bf16(v16bf a, v16bf b, v8f c) {
  // D = A(16x32) * B(32x16) + C, fp32 accumulate
  return __builtin_amdgcn_wmma_f32_16x16x32_bf16(false, a, false, b, (short)0, c,
                                                 false, false);
}

// Load one 16-lane-striped operand fragment.  Memory layout is row-major with
// K contiguous, so per the CDNA5 16-bit A/B VGPR layout each lane needs two
// contiguous 8-element (16B) chunks: K = k + half*8 and K = k + 16 + half*8.
__device__ __forceinline__ v16bf load_frag(const bf16_t* base, int ld, int row,
                                           int k, int half) {
  const bf16_t* p = base + (size_t)row * ld + k + half * 8;
  v8bf lo = *(const v8bf*)(p);
  v8bf hi = *(const v8bf*)(p + 16);
  v16bf r;
#pragma unroll
  for (int i = 0; i < 8; ++i) { r[i] = lo[i]; r[i + 8] = hi[i]; }
  return r;
}

// ---------------------------------------------------------------------------
// Prep: transpose+convert [F][N] fp32 -> [N][F] bf16  (per batch in grid.z)
// ---------------------------------------------------------------------------
__global__ void k_transpose_cvt(const float* __restrict__ src,
                                bf16_t* __restrict__ dst, int F, int N) {
  int b = blockIdx.z;
  int idx = blockIdx.x * blockDim.x + threadIdx.x;
  if (idx >= N * F) return;
  int n = idx / F, f = idx % F;
  dst[(size_t)b * N * F + idx] = (bf16_t)src[(size_t)b * F * N + (size_t)f * N + n];
}

// Prep: convert fp32 [R][C] (row stride srcLd, col offset) -> bf16 [Rp][Cp], 0-pad
__global__ void k_cvt_pad(const float* __restrict__ src, bf16_t* __restrict__ dst,
                          int R, int C, int Rp, int Cp, int srcLd, int colOff) {
  int idx = blockIdx.x * blockDim.x + threadIdx.x;
  if (idx >= Rp * Cp) return;
  int r = idx / Cp, c = idx % Cp;
  float v = (r < R && c < C) ? src[(size_t)r * srcLd + colOff + c] : 0.f;
  dst[idx] = (bf16_t)v;
}

// ---------------------------------------------------------------------------
// Generic bf16 WMMA GEMM:  C[b][m][n] = sum_k A[b][m][k] * BT[n][k] (+bias, relu)
// One wave per 16x16 output tile. A,BT row-major, K contiguous, K % 32 == 0.
// ---------------------------------------------------------------------------
__global__ void k_gemm_bf16(const bf16_t* __restrict__ A, long long sA, int lda,
                            const bf16_t* __restrict__ BT, long long sB, int ldb,
                            float* __restrict__ Cf, bf16_t* __restrict__ Cb,
                            long long sC, int ldc,
                            const float* __restrict__ bias, int K, int relu) {
  const int lane = threadIdx.x & 31;
  const int half = lane >> 4, lr = lane & 15;
  const int m0 = blockIdx.x * 16, n0 = blockIdx.y * 16, b = blockIdx.z;
  const bf16_t* Ab = A + (size_t)b * sA;
  const bf16_t* Bb = BT + (size_t)b * sB;

  v8f acc = {0.f, 0.f, 0.f, 0.f, 0.f, 0.f, 0.f, 0.f};
  for (int k = 0; k < K; k += 32) {
    v16bf af = load_frag(Ab, lda, m0 + lr, k, half);
    v16bf bf = load_frag(Bb, ldb, n0 + lr, k, half);
    acc = wmma_bf16(af, bf, acc);
  }
  const float bv = bias ? bias[n0 + lr] : 0.f;
#pragma unroll
  for (int v = 0; v < 8; ++v) {
    float y = acc[v] + bv;
    if (relu) y = fmaxf(y, 0.f);
    int row = m0 + v + 8 * half, col = n0 + lr;
    size_t idx = (size_t)b * sC + (size_t)row * ldc + col;
    if (Cf) Cf[idx] = y;
    if (Cb) Cb[idx] = (bf16_t)y;
  }
}

// ---------------------------------------------------------------------------
// Fused pair-MLP tail. One workgroup (256 thr = 8 waves) per (nb,mb,b) tile.
// Builds X[pair=256][H1K] bf16 in LDS, GEMMs against W2bf [H2P][H1K],
// fuses bias+relu+dot(w3) into the epilogue, softplus-combines v1/v2 -> R.
// ---------------------------------------------------------------------------
#define PAIR_SMEM (N_ * H1K * 2 + 2 * N_ * 4)

__global__ void k_pair_tail(const float* __restrict__ a1, const float* __restrict__ a2,
                            const float* __restrict__ b1, const float* __restrict__ b2,
                            const float* __restrict__ kb1,
                            const bf16_t* __restrict__ W2bf,
                            const float* __restrict__ kb2,
                            const float* __restrict__ kw3,
                            const float* __restrict__ kb3,
                            float* __restrict__ R) {
  extern __shared__ char smem[];
  bf16_t* X    = (bf16_t*)smem;                 // [256][H1K]
  float*  vbuf = (float*)(smem + N_ * H1K * 2); // [2][256]

  const int t = threadIdx.x;                    // pair index 0..255
  const int nb = blockIdx.x, mb = blockIdx.y, b = blockIdx.z;
  const int nloc = t >> 4, mloc = t & 15;
  const int n = nb * 16 + nloc, m = mb * 16 + mloc;
  const int wid = t >> 5, lane = t & 31, half = lane >> 4, lr = lane & 15;

  // per-lane epilogue constants (cols this lane touches: nt*16 + lr)
  float w3v[7], b2v[7];
#pragma unroll
  for (int nt = 0; nt < 7; ++nt) {
    int col = nt * 16 + lr;
    w3v[nt] = (col < 100) ? kw3[col] : 0.f;
    b2v[nt] = (col < 100) ? kb2[col] : 0.f;
  }

  const float* rowN[2], * rowM[2];
  rowN[0] = a1 + ((size_t)b * N_ + n) * H1P;  // v1: a1[n] + b2[m]
  rowM[0] = b2 + ((size_t)b * N_ + m) * H1P;
  rowN[1] = a2 + ((size_t)b * N_ + n) * H1P;  // v2: b1[m] + a2[n]
  rowM[1] = b1 + ((size_t)b * N_ + m) * H1P;

  for (int pass = 0; pass < 2; ++pass) {
    // ---- build X (one row per thread, fp32 add -> bf16) ----
    bf16_t* xr = X + (size_t)t * H1K;
    const float* pn = rowN[pass];
    const float* pm = rowM[pass];
    for (int h = 0; h < 300; ++h) xr[h] = (bf16_t)(pn[h] + pm[h] + kb1[h]);
    for (int h = 300; h < H1K; ++h) xr[h] = (bf16_t)0.f;
    __syncthreads();

    // ---- GEMM: wave owns M-tiles {2*wid, 2*wid+1} x all 7 N-tiles ----
    v8f acc[2][7];
    const v8f zero = {0.f, 0.f, 0.f, 0.f, 0.f, 0.f, 0.f, 0.f};
#pragma unroll
    for (int mt = 0; mt < 2; ++mt)
#pragma unroll
      for (int nt = 0; nt < 7; ++nt) acc[mt][nt] = zero;

    for (int k = 0; k < H1K; k += 32) {
      v16bf af0 = load_frag(X, H1K, (wid * 2 + 0) * 16 + lr, k, half);
      v16bf af1 = load_frag(X, H1K, (wid * 2 + 1) * 16 + lr, k, half);
#pragma unroll
      for (int nt = 0; nt < 7; ++nt) {
        v16bf bf = load_frag(W2bf, H1K, nt * 16 + lr, k, half);
        acc[0][nt] = wmma_bf16(af0, bf, acc[0][nt]);
        acc[1][nt] = wmma_bf16(af1, bf, acc[1][nt]);
      }
    }

    // ---- epilogue: relu(acc + kb2) . w3, reduce across the 16 N-lanes ----
#pragma unroll
    for (int mt = 0; mt < 2; ++mt) {
      float p[8] = {0.f, 0.f, 0.f, 0.f, 0.f, 0.f, 0.f, 0.f};
#pragma unroll
      for (int nt = 0; nt < 7; ++nt)
#pragma unroll
        for (int v = 0; v < 8; ++v) {
          float y = acc[mt][nt][v] + b2v[nt];
          y = fmaxf(y, 0.f);
          p[v] += y * w3v[nt];
        }
#pragma unroll
      for (int off = 1; off < 16; off <<= 1)
#pragma unroll
        for (int v = 0; v < 8; ++v) p[v] += __shfl_xor(p[v], off, 32);
      if (lr == 0) {
        int rowbase = (wid * 2 + mt) * 16 + 8 * half;
#pragma unroll
        for (int v = 0; v < 8; ++v) vbuf[pass * N_ + rowbase + v] = p[v];
      }
    }
    __syncthreads();
  }

  // ---- combine + softplus, write R[b][n][m] ----
  float vsum = 0.5f * (vbuf[t] + vbuf[N_ + t]) + kb3[0];
  float r = fmaxf(vsum, 0.f) + log1pf(expf(-fabsf(vsum)));
  R[((size_t)b * N_ + n) * N_ + m] = r;
}

// ---------------------------------------------------------------------------
// Multilateration stats: per batch compute Pbar and inv(M) (3x3 symmetric).
// stats[b] = {Pbar(3), i00,i01,i02,i11,i12,i22}
// ---------------------------------------------------------------------------
__global__ void k_mstats(const float* __restrict__ anchor_pts, float* __restrict__ stats) {
  const int b = blockIdx.x, lane = threadIdx.x;  // 32 threads
  const float* P = anchor_pts + (size_t)b * 3 * N_;
  float sx = 0.f, sy = 0.f, sz = 0.f;
  for (int n = lane; n < N_; n += 32) { sx += P[n]; sy += P[N_ + n]; sz += P[2 * N_ + n]; }
  for (int off = 16; off; off >>= 1) {
    sx += __shfl_xor(sx, off, 32); sy += __shfl_xor(sy, off, 32); sz += __shfl_xor(sz, off, 32);
  }
  const float bx = sx / N_, by = sy / N_, bz = sz / N_;
  float m00 = 0.f, m01 = 0.f, m02 = 0.f, m11 = 0.f, m12 = 0.f, m22 = 0.f;
  for (int n = lane; n < N_; n += 32) {
    float ax = P[n] - bx, ay = P[N_ + n] - by, az = P[2 * N_ + n] - bz;
    m00 += ax * ax; m01 += ax * ay; m02 += ax * az;
    m11 += ay * ay; m12 += ay * az; m22 += az * az;
  }
  for (int off = 16; off; off >>= 1) {
    m00 += __shfl_xor(m00, off, 32); m01 += __shfl_xor(m01, off, 32);
    m02 += __shfl_xor(m02, off, 32); m11 += __shfl_xor(m11, off, 32);
    m12 += __shfl_xor(m12, off, 32); m22 += __shfl_xor(m22, off, 32);
  }
  if (lane == 0) {
    m00 /= N_; m01 /= N_; m02 /= N_; m11 /= N_; m12 /= N_; m22 /= N_;
    float c00 = m11 * m22 - m12 * m12;
    float c01 = m02 * m12 - m01 * m22;
    float c02 = m01 * m12 - m02 * m11;
    float det = m00 * c00 + m01 * c01 + m02 * c02;
    float inv = 1.f / det;
    float* s = stats + b * 12;
    s[0] = bx; s[1] = by; s[2] = bz;
    s[3] = c00 * inv;                       // i00
    s[4] = c01 * inv;                       // i01
    s[5] = c02 * inv;                       // i02
    s[6] = (m00 * m22 - m02 * m02) * inv;   // i11
    s[7] = (m02 * m01 - m00 * m12) * inv;   // i12
    s[8] = (m00 * m11 - m01 * m01) * inv;   // i22
  }
}

// ---------------------------------------------------------------------------
// Per (b, a): v = 0.5/N * sum_n (|P_B|^2 - R^2) * (P_B - Pbar); pred = Minv v.
// Writes full_flow[0..2], flow, corr_points directly into d_out.
// ---------------------------------------------------------------------------
__global__ void k_solve(const float* __restrict__ anchor_pts,
                        const float* __restrict__ action_pts,
                        const float* __restrict__ R,
                        const float* __restrict__ stats,
                        float* __restrict__ dout) {
  int idx = blockIdx.x * blockDim.x + threadIdx.x;
  if (idx >= B_ * N_) return;
  const int b = idx >> 8, a = idx & 255;
  const float* s = stats + b * 12;
  const float bx = s[0], by = s[1], bz = s[2];
  const float* P = anchor_pts + (size_t)b * 3 * N_;
  const float* Rr = R + ((size_t)b * N_ + a) * N_;
  float v0 = 0.f, v1 = 0.f, v2 = 0.f;
  for (int n = 0; n < N_; ++n) {
    float px = P[n], py = P[N_ + n], pz = P[2 * N_ + n];
    float r = Rr[n];
    float c = px * px + py * py + pz * pz - r * r;
    v0 += c * (px - bx); v1 += c * (py - by); v2 += c * (pz - bz);
  }
  const float sc = 0.5f / (float)N_;
  v0 *= sc; v1 *= sc; v2 *= sc;
  float pred[3];
  pred[0] = s[3] * v0 + s[4] * v1 + s[5] * v2;
  pred[1] = s[4] * v0 + s[6] * v1 + s[7] * v2;
  pred[2] = s[5] * v0 + s[7] * v1 + s[8] * v2;
#pragma unroll
  for (int d = 0; d < 3; ++d) {
    float fl = pred[d] - action_pts[(size_t)b * 3 * N_ + d * N_ + a];
    dout[((size_t)b * 4 + d) * N_ + a] = fl;                  // full_flow c=0..2
    dout[2048 + ((size_t)b * 3 + d) * N_ + a] = fl;           // flow
    dout[3584 + ((size_t)b * 3 + d) * N_ + a] = pred[d];      // corr_points
  }
}

// ---------------------------------------------------------------------------
// PointNet head: weight[b][n] = head_w . act5[b][n][:]  -> full_flow channel 3
// ---------------------------------------------------------------------------
__global__ void k_head(const bf16_t* __restrict__ act5,
                       const float* __restrict__ head_w,
                       float* __restrict__ dout) {
  int idx = blockIdx.x * blockDim.x + threadIdx.x;
  if (idx >= B_ * N_) return;
  const int b = idx >> 8, n = idx & 255;
  const bf16_t* x = act5 + ((size_t)b * N_ + n) * 512;
  float sacc = 0.f;
  for (int c = 0; c < 512; ++c) sacc += (float)x[c] * head_w[c];
  dout[((size_t)b * 4 + 3) * N_ + n] = sacc;
}

// ---------------------------------------------------------------------------
extern "C" void kernel_launch(void* const* d_in, const int* in_sizes, int n_in,
                              void* d_out, int out_size, void* d_ws, size_t ws_size,
                              hipStream_t stream) {
  const float* action_emb = (const float*)d_in[0];   // [2,512,256]
  const float* anchor_emb = (const float*)d_in[1];   // [2,512,256]
  const float* action_pts = (const float*)d_in[2];   // [2,3,256]
  const float* anchor_pts = (const float*)d_in[3];   // [2,3,256]
  const float* kw1 = (const float*)d_in[4];          // [300,1024]
  const float* kb1 = (const float*)d_in[5];          // [300]
  const float* kw2 = (const float*)d_in[6];          // [100,300]
  const float* kb2 = (const float*)d_in[7];          // [100]
  const float* kw3 = (const float*)d_in[8];          // [1,100]
  const float* kb3 = (const float*)d_in[9];          // [1]
  const float* pnw[5] = {(const float*)d_in[10], (const float*)d_in[12],
                         (const float*)d_in[14], (const float*)d_in[16],
                         (const float*)d_in[18]};
  const float* pnb[5] = {(const float*)d_in[11], (const float*)d_in[13],
                         (const float*)d_in[15], (const float*)d_in[17],
                         (const float*)d_in[19]};
  const float* head_w = (const float*)d_in[20];      // [1,512]
  float* dout = (float*)d_out;

  // ---- workspace carve-out (256B aligned) ----
  char* ws = (char*)d_ws;
  size_t off = 0;
  auto take = [&](size_t bytes) { size_t r = off; off += (bytes + 255) & ~(size_t)255; return r; };
  bf16_t* embT_A = (bf16_t*)(ws + take((size_t)B_ * N_ * F_ * 2));
  bf16_t* embT_B = (bf16_t*)(ws + take((size_t)B_ * N_ * F_ * 2));
  bf16_t* W1a    = (bf16_t*)(ws + take((size_t)H1P * F_ * 2));
  bf16_t* W1b    = (bf16_t*)(ws + take((size_t)H1P * F_ * 2));
  bf16_t* W2bf   = (bf16_t*)(ws + take((size_t)H2P * H1K * 2));
  bf16_t* pw0    = (bf16_t*)(ws + take((size_t)64 * 512 * 2));
  bf16_t* pw1    = (bf16_t*)(ws + take((size_t)64 * 64 * 2));
  bf16_t* pw2    = (bf16_t*)(ws + take((size_t)64 * 64 * 2));
  bf16_t* pw3    = (bf16_t*)(ws + take((size_t)128 * 64 * 2));
  bf16_t* pw4    = (bf16_t*)(ws + take((size_t)512 * 128 * 2));
  float*  a1     = (float*)(ws + take((size_t)B_ * N_ * H1P * 4));
  float*  a2     = (float*)(ws + take((size_t)B_ * N_ * H1P * 4));
  float*  b1     = (float*)(ws + take((size_t)B_ * N_ * H1P * 4));
  float*  b2     = (float*)(ws + take((size_t)B_ * N_ * H1P * 4));
  bf16_t* act1   = (bf16_t*)(ws + take((size_t)B_ * N_ * 64 * 2));
  bf16_t* act2   = (bf16_t*)(ws + take((size_t)B_ * N_ * 64 * 2));
  bf16_t* act3   = (bf16_t*)(ws + take((size_t)B_ * N_ * 64 * 2));
  bf16_t* act4   = (bf16_t*)(ws + take((size_t)B_ * N_ * 128 * 2));
  bf16_t* act5   = (bf16_t*)(ws + take((size_t)B_ * N_ * 512 * 2));
  float*  Rbuf   = (float*)(ws + take((size_t)B_ * N_ * N_ * 4));
  float*  stats  = (float*)(ws + take((size_t)B_ * 12 * 4));
  (void)ws_size; (void)in_sizes; (void)n_in; (void)out_size;

  // ---- prep: bf16 conversions ----
  {
    dim3 g((N_ * F_ + 255) / 256, 1, B_);
    k_transpose_cvt<<<g, 256, 0, stream>>>(action_emb, embT_A, F_, N_);
    k_transpose_cvt<<<g, 256, 0, stream>>>(anchor_emb, embT_B, F_, N_);
  }
  auto cvt = [&](const float* s, bf16_t* d, int R, int C, int Rp, int Cp, int ld, int co) {
    k_cvt_pad<<<dim3((Rp * Cp + 255) / 256), 256, 0, stream>>>(s, d, R, C, Rp, Cp, ld, co);
  };
  cvt(kw1, W1a, 300, 512, H1P, 512, 1024, 0);
  cvt(kw1, W1b, 300, 512, H1P, 512, 1024, 512);
  cvt(kw2, W2bf, 100, 300, H2P, H1K, 300, 0);
  cvt(pnw[0], pw0, 64, 512, 64, 512, 512, 0);
  cvt(pnw[1], pw1, 64, 64, 64, 64, 64, 0);
  cvt(pnw[2], pw2, 64, 64, 64, 64, 64, 0);
  cvt(pnw[3], pw3, 128, 64, 128, 64, 64, 0);
  cvt(pnw[4], pw4, 512, 128, 512, 128, 128, 0);

  // ---- layer-1 GEMMs: [n][h] = embT[b] (A) x W1{a,b} (BT), fp32 out ----
  const long long sEmb = (long long)N_ * F_, sA1 = (long long)N_ * H1P;
  k_gemm_bf16<<<dim3(N_ / 16, H1P / 16, B_), 32, 0, stream>>>(
      embT_A, sEmb, F_, W1a, 0, F_, a1, nullptr, sA1, H1P, nullptr, F_, 0);
  k_gemm_bf16<<<dim3(N_ / 16, H1P / 16, B_), 32, 0, stream>>>(
      embT_A, sEmb, F_, W1b, 0, F_, a2, nullptr, sA1, H1P, nullptr, F_, 0);
  k_gemm_bf16<<<dim3(N_ / 16, H1P / 16, B_), 32, 0, stream>>>(
      embT_B, sEmb, F_, W1a, 0, F_, b1, nullptr, sA1, H1P, nullptr, F_, 0);
  k_gemm_bf16<<<dim3(N_ / 16, H1P / 16, B_), 32, 0, stream>>>(
      embT_B, sEmb, F_, W1b, 0, F_, b2, nullptr, sA1, H1P, nullptr, F_, 0);

  // ---- PointNet layers (bias+relu fused, bf16 activations, k-contiguous) ----
  k_gemm_bf16<<<dim3(N_ / 16, 64 / 16, B_), 32, 0, stream>>>(
      embT_A, sEmb, F_, pw0, 0, 512, nullptr, act1, (long long)N_ * 64, 64, pnb[0], 512, 1);
  k_gemm_bf16<<<dim3(N_ / 16, 64 / 16, B_), 32, 0, stream>>>(
      act1, (long long)N_ * 64, 64, pw1, 0, 64, nullptr, act2, (long long)N_ * 64, 64, pnb[1], 64, 1);
  k_gemm_bf16<<<dim3(N_ / 16, 64 / 16, B_), 32, 0, stream>>>(
      act2, (long long)N_ * 64, 64, pw2, 0, 64, nullptr, act3, (long long)N_ * 64, 64, pnb[2], 64, 1);
  k_gemm_bf16<<<dim3(N_ / 16, 128 / 16, B_), 32, 0, stream>>>(
      act3, (long long)N_ * 64, 64, pw3, 0, 64, nullptr, act4, (long long)N_ * 128, 128, pnb[3], 64, 1);
  k_gemm_bf16<<<dim3(N_ / 16, 512 / 16, B_), 32, 0, stream>>>(
      act4, (long long)N_ * 128, 128, pw4, 0, 128, nullptr, act5, (long long)N_ * 512, 512, pnb[4], 128, 1);

  // ---- fused pair-MLP tail -> R_est ----
  k_pair_tail<<<dim3(N_ / 16, N_ / 16, B_), 256, PAIR_SMEM, stream>>>(
      a1, a2, b1, b2, kb1, W2bf, kb2, kw3, kb3, Rbuf);

  // ---- multilateration + outputs ----
  k_mstats<<<dim3(B_), 32, 0, stream>>>(anchor_pts, stats);
  k_solve<<<dim3((B_ * N_ + 255) / 256), 256, 0, stream>>>(
      anchor_pts, action_pts, Rbuf, stats, dout);
  k_head<<<dim3((B_ * N_ + 255) / 256), 256, 0, stream>>>(act5, head_w, dout);
}
